// BiMambaEncoder_2473901162710
// MI455X (gfx1250) — compile-verified, matrix operand order checked by reading, not run on hardware
//
#include <hip/hip_runtime.h>
#include <math.h>

// ---------------- problem constants (from the reference) ----------------
#define D_MODEL 512
#define D_INNER 1024
#define DT_RANK 32
#define D_STATE 16
#define KCONV   4
#define BATCH   2
#define SEQ     2048
#define NTOK    (BATCH * SEQ)            // 4096 tokens
#define XPROJ_N (DT_RANK + 2 * D_STATE)  // 64

typedef __attribute__((ext_vector_type(16))) _Float16 v16h;
typedef __attribute__((ext_vector_type(8)))  _Float16 v8h;
typedef __attribute__((ext_vector_type(8)))  float    v8f;

enum { ACT_NONE = 0, ACT_GELU = 1, ACT_SOFTPLUS = 2 };

// LDS layout per buffer (double-buffered dynamic shared memory; byte offsets
// are workgroup-relative LDS addresses usable by the async-copy instructions):
//   As: 128 rows x 40 halves (row stride 80B)  -> bytes [0, 10240)
//   Bs:  64 rows x 40 halves                   -> bytes [10240, 15360)
#define A_ROW_B   80
#define A_ROW_H   40
#define BS_BASE_B 10240
#define BUF_B     15360
#define SMEM_B    (2 * BUF_B)           // two pipeline buffers

template<int ACT>
__device__ __forceinline__ float apply_act(float v) {
  if (ACT == ACT_GELU)     return 0.5f * v * (1.0f + erff(v * 0.70710678118654752f));
  if (ACT == ACT_SOFTPLUS) return (v > 20.0f) ? v : log1pf(__expf(v));
  return v;
}

// one 16-byte async copy global -> LDS (ASYNCcnt-tracked, no VGPR staging)
__device__ __forceinline__ void async_cp16(unsigned lds_byte, const _Float16* g) {
  asm volatile("global_load_async_to_lds_b128 %0, %1, off"
               :: "v"(lds_byte), "v"((unsigned long long)(const void*)g)
               : "memory");
}
__device__ __forceinline__ void async_wait0() {
  asm volatile("s_wait_asynccnt 0x0" ::: "memory");
}

// stage one K-step (A 128x32 + B 64x32, f16) into LDS buffer at byte `base`
__device__ __forceinline__ void stage_tiles(unsigned base,
                                            const _Float16* __restrict__ A, int lda,
                                            const _Float16* __restrict__ W, int K,
                                            int row0, int col0, int k0, int tid) {
#pragma unroll
  for (int i = 0; i < 2; ++i) {         // A: 512 x 16B chunks
    int c  = tid + i * 256;
    int rl = c >> 2;
    int ch = c & 3;
    async_cp16(base + (unsigned)(rl * A_ROW_B + ch * 16),
               A + (size_t)(row0 + rl) * lda + k0 + ch * 8);
  }
  {                                      // B: 256 x 16B chunks
    int rl = tid >> 2;
    int ch = tid & 3;
    async_cp16(base + (unsigned)(BS_BASE_B + rl * A_ROW_B + ch * 16),
               W + (size_t)(col0 + rl) * K + k0 + ch * 8);
  }
}

// =======================================================================
// WMMA GEMM:  C[M,N] = ACT( A[M,K] @ W[N,K]^T (+ bias) ) (+ resid)
// f16 operands, fp32 accumulate (v_wmma_f32_16x16x32_f16).
// Block tile 128x64, 8 waves, each wave 32x32 (2x2 WMMA tiles).
// Double-buffered async-to-LDS pipeline: tile k+1's DMA is issued before
// computing tile k, so the async engine runs under the WMMAs.
// Requires: M % 128 == 0, N % 64 == 0, K % 32 == 0 (true for all calls).
// =======================================================================
template<int ACT, bool HAS_BIAS, bool HAS_RES, bool OUT_F, bool OUT_H>
__global__ void __launch_bounds__(256)
gemm_wmma_kernel(const _Float16* __restrict__ A, int lda,
                 const _Float16* __restrict__ W,   // N x K row-major, f16
                 const float* __restrict__ bias,
                 const float* __restrict__ resid, int ldres,
                 float* __restrict__ Cf, _Float16* __restrict__ Ch,
                 int ldc, int K)
{
  extern __shared__ _Float16 smem[];   // LDS byte 0 for this workgroup

  const int tid  = threadIdx.x;
  const int lane = tid & 31;
  const int wave = tid >> 5;
  const int mwb  = (wave & 3) * 32;
  const int nwb  = (wave >> 2) * 32;
  const int row0 = blockIdx.y * 128;
  const int col0 = blockIdx.x * 64;

  const int r    = lane & 15;
  const int hi   = lane >> 4;
  const int kOff = hi ? 8 : 0;         // ISA f16 A/B fragment lane layout

  v8f acc[2][2] = {};

  const int nk = K >> 5;
  stage_tiles(0u, A, lda, W, K, row0, col0, 0, tid);

  for (int kt = 0; kt < nk; ++kt) {
    async_wait0();        // this wave's outstanding copies (tile kt) landed
    __syncthreads();      // all waves: tile kt visible, prev buffer free
    const unsigned cur = (kt & 1) ? BUF_B : 0u;
    if (kt + 1 < nk)      // prefetch tile kt+1 into the other buffer
      stage_tiles(cur ^ BUF_B, A, lda, W, K, row0, col0, (kt + 1) << 5, tid);

    const _Float16* sm = smem + (cur >> 1);    // halves offset
    v16h af[2], bf[2];
#pragma unroll
    for (int i = 0; i < 2; ++i) {
      const _Float16* p = sm + (mwb + 16 * i + r) * A_ROW_H;
      v8h lo = *reinterpret_cast<const v8h*>(p + kOff);
      v8h hh = *reinterpret_cast<const v8h*>(p + 16 + kOff);
      af[i] = __builtin_shufflevector(lo, hh, 0,1,2,3,4,5,6,7,8,9,10,11,12,13,14,15);
    }
#pragma unroll
    for (int j = 0; j < 2; ++j) {
      const _Float16* p = sm + (BS_BASE_B / 2) + (nwb + 16 * j + r) * A_ROW_H;
      v8h lo = *reinterpret_cast<const v8h*>(p + kOff);
      v8h hh = *reinterpret_cast<const v8h*>(p + 16 + kOff);
      bf[j] = __builtin_shufflevector(lo, hh, 0,1,2,3,4,5,6,7,8,9,10,11,12,13,14,15);
    }
#pragma unroll
    for (int i = 0; i < 2; ++i)
#pragma unroll
      for (int j = 0; j < 2; ++j)
        acc[i][j] = __builtin_amdgcn_wmma_f32_16x16x32_f16(
            false, af[i], false, bf[j], (short)0, acc[i][j], false, false);
  }

  // ---- branch-free epilogue (configuration is compile-time) ----
#pragma unroll
  for (int j = 0; j < 2; ++j) {
    int col = col0 + nwb + 16 * j + r;
    float bv = HAS_BIAS ? bias[col] : 0.0f;
#pragma unroll
    for (int i = 0; i < 2; ++i) {
#pragma unroll
      for (int p = 0; p < 8; ++p) {
        int row = row0 + mwb + 16 * i + p + hi * 8; // C layout: VGPR p -> M=p / p+8
        float v = acc[i][j][p] + bv;
        v = apply_act<ACT>(v);
        if (HAS_RES) v += resid[(size_t)row * ldres + col];
        if (OUT_F)  Cf[(size_t)row * ldc + col] = v;
        if (OUT_H)  Ch[(size_t)row * ldc + col] = (_Float16)v;
      }
    }
  }
}

// =======================================================================
// fp32 -> f16 conversion (weights, once per launch)
// =======================================================================
__global__ void __launch_bounds__(256)
cvt_f16_kernel(const float* __restrict__ src, _Float16* __restrict__ dst, int n)
{
  int i = blockIdx.x * 256 + threadIdx.x;
  if (i < n) dst[i] = (_Float16)src[i];
}

// =======================================================================
// LayerNorm over D_MODEL=512; optional second operand summed in, with
// time-flip indexing (out_f + flip(out_b)). f32 and/or f16 outputs.
// One 256-thread block per token; each thread owns 2 elements.
// =======================================================================
__global__ void __launch_bounds__(256)
ln_kernel(const float* __restrict__ a, const float* __restrict__ b2, int flip,
          const float* __restrict__ g, const float* __restrict__ be,
          float* __restrict__ out, _Float16* __restrict__ out_h)
{
  __shared__ float red[256];
  const int token = blockIdx.x;
  const int bat = token / SEQ, t = token % SEQ;
  const int tid = threadIdx.x;

  const float* pa = a + (size_t)token * D_MODEL;
  float v0 = pa[tid], v1 = pa[tid + 256];
  if (b2) {
    int t2 = flip ? (SEQ - 1 - t) : t;
    const float* pb = b2 + ((size_t)bat * SEQ + t2) * D_MODEL;
    v0 += pb[tid]; v1 += pb[tid + 256];
  }

  red[tid] = v0 + v1;
  __syncthreads();
  for (int off = 128; off > 0; off >>= 1) {
    if (tid < off) red[tid] += red[tid + off];
    __syncthreads();
  }
  const float mean = red[0] * (1.0f / D_MODEL);
  __syncthreads();

  const float d0 = v0 - mean, d1 = v1 - mean;
  red[tid] = d0 * d0 + d1 * d1;
  __syncthreads();
  for (int off = 128; off > 0; off >>= 1) {
    if (tid < off) red[tid] += red[tid + off];
    __syncthreads();
  }
  const float rstd = rsqrtf(red[0] * (1.0f / D_MODEL) + 1e-5f);

  const float r0 = d0 * rstd * g[tid]       + be[tid];
  const float r1 = d1 * rstd * g[tid + 256] + be[tid + 256];
  if (out) {
    float* po = out + (size_t)token * D_MODEL;
    po[tid] = r0; po[tid + 256] = r1;
  }
  if (out_h) {
    _Float16* ph = out_h + (size_t)token * D_MODEL;
    ph[tid] = (_Float16)r0; ph[tid + 256] = (_Float16)r1;
  }
}

// =======================================================================
// Depthwise causal conv (K=4) + bias + SiLU, direction-aware.
// =======================================================================
__global__ void __launch_bounds__(256)
conv_silu_kernel(const float* __restrict__ xz, const float* __restrict__ w,
                 const float* __restrict__ cb,
                 float* __restrict__ xs, _Float16* __restrict__ xs_h, int dir)
{
  size_t idx = (size_t)blockIdx.x * 256 + threadIdx.x;   // B*S*D_INNER
  int d   = (int)(idx % D_INNER);
  int tau = (int)((idx / D_INNER) % SEQ);
  int bat = (int)(idx / ((size_t)D_INNER * SEQ));

  float acc = cb[d];
#pragma unroll
  for (int j = 0; j < KCONV; ++j) {
    int ti = tau - (KCONV - 1) + j;
    if (ti >= 0) {
      int tg = dir ? (SEQ - 1 - ti) : ti;
      acc += w[d * KCONV + j] *
             xz[((size_t)bat * SEQ + tg) * (2 * D_INNER) + d];
    }
  }
  const float s = acc / (1.0f + __expf(-acc));   // silu
  xs[idx]   = s;
  xs_h[idx] = (_Float16)s;
}

// =======================================================================
// Selective scan + D-skip + silu(z) gate, fused.
// One thread per (batch, channel): 16-state recurrence in registers.
// =======================================================================
__global__ void __launch_bounds__(256)
scan_gate_kernel(const float* __restrict__ xs, const float* __restrict__ dbl,
                 const float* __restrict__ delta, const float* __restrict__ xz,
                 const float* __restrict__ A_log, const float* __restrict__ Dsk,
                 _Float16* __restrict__ yg, int dir)
{
  const int tg0 = blockIdx.x * 256 + threadIdx.x;   // 0 .. B*D_INNER-1
  const int d   = tg0 % D_INNER;
  const int bat = tg0 / D_INNER;

  float Aneg[D_STATE], h[D_STATE];
#pragma unroll
  for (int s = 0; s < D_STATE; ++s) {
    Aneg[s] = -__expf(A_log[d * D_STATE + s]);
    h[s] = 0.0f;
  }
  const float Dd = Dsk[d];

  for (int tau = 0; tau < SEQ; ++tau) {
    const size_t base = (size_t)bat * SEQ + tau;
    const float del = delta[base * D_INNER + d];
    const float x   = xs[base * D_INNER + d];
    const float* Brow = dbl + base * XPROJ_N + DT_RANK;
    const float* Crow = Brow + D_STATE;
    const float dx = del * x;
    float y = 0.0f;
#pragma unroll
    for (int s = 0; s < D_STATE; ++s) {
      float dA = __expf(del * Aneg[s]);
      h[s] = dA * h[s] + dx * Brow[s];
      y += h[s] * Crow[s];
    }
    const int tgl = dir ? (SEQ - 1 - tau) : tau;
    const float z = xz[((size_t)bat * SEQ + tgl) * (2 * D_INNER) + D_INNER + d];
    const float sz = z / (1.0f + __expf(-z));
    yg[base * D_INNER + d] = (_Float16)((y + Dd * x) * sz);
  }
}

// =======================================================================
extern "C" void kernel_launch(void* const* d_in, const int* in_sizes, int n_in,
                              void* d_out, int out_size, void* d_ws, size_t ws_size,
                              hipStream_t stream)
{
  (void)in_sizes; (void)n_in; (void)out_size; (void)ws_size;
  const float* x        = (const float*)d_in[0];
  const float* n1w      = (const float*)d_in[1];
  const float* n1b      = (const float*)d_in[2];
  const float* n2w      = (const float*)d_in[3];
  const float* n2b      = (const float*)d_in[4];
  const float* in_proj  = (const float*)d_in[5];
  const float* conv_w   = (const float*)d_in[6];
  const float* conv_b   = (const float*)d_in[7];
  const float* x_proj   = (const float*)d_in[8];
  const float* dt_proj  = (const float*)d_in[9];
  const float* dt_b     = (const float*)d_in[10];
  const float* A_log    = (const float*)d_in[11];
  const float* Dsk      = (const float*)d_in[12];
  const float* out_proj = (const float*)d_in[13];
  const float* ff_w1    = (const float*)d_in[14];
  const float* ff_b1    = (const float*)d_in[15];
  const float* ff_w2    = (const float*)d_in[16];
  const float* ff_b2    = (const float*)d_in[17];
  float* out = (float*)d_out;

  // ---------------- workspace layout ----------------
  float* ws = (float*)d_ws;
  size_t off = 0;
  float* xz   = ws + off; off += (size_t)NTOK * 2 * D_INNER;  // f32: conv/scan
  float* xs   = ws + off; off += (size_t)NTOK * D_INNER;      // f32: scan
  float* dbl  = ws + off; off += (size_t)NTOK * XPROJ_N;      // f32: scan B/C
  float* del  = ws + off; off += (size_t)NTOK * D_INNER;      // f32: scan
  float* outf = ws + off; off += (size_t)NTOK * D_MODEL;
  float* outb = ws + off; off += (size_t)NTOK * D_MODEL;

  _Float16* hb = (_Float16*)(ws + off);
  size_t ho = 0;
  _Float16* inproj_h  = hb + ho; ho += (size_t)2 * D_INNER * D_MODEL;
  _Float16* xproj_h   = hb + ho; ho += (size_t)XPROJ_N * D_INNER;
  _Float16* dtproj_h  = hb + ho; ho += (size_t)D_INNER * DT_RANK;
  _Float16* outproj_h = hb + ho; ho += (size_t)D_MODEL * D_INNER;
  _Float16* ffw1_h    = hb + ho; ho += (size_t)4 * D_MODEL * D_MODEL;
  _Float16* ffw2_h    = hb + ho; ho += (size_t)D_MODEL * 4 * D_MODEL;
  _Float16* xn_h      = hb + ho; ho += (size_t)NTOK * D_MODEL;
  _Float16* xs_h      = hb + ho; ho += (size_t)NTOK * D_INNER;
  _Float16* dbl_h     = hb + ho; ho += (size_t)NTOK * XPROJ_N;
  _Float16* yg_h      = hb + ho; ho += (size_t)NTOK * D_INNER;
  _Float16* ffb_h     = hb + ho; ho += (size_t)NTOK * 4 * D_MODEL;
  _Float16* m_h       = xn_h;   // xn_h dead after in_proj

  const dim3 blk(256);
  const int mt = NTOK / 128;   // 32 M-tiles

  // ---- convert weights to f16 once per launch ----
  struct { const float* s; _Float16* d; int n; } cv[6] = {
    { in_proj,  inproj_h,  2 * D_INNER * D_MODEL },
    { x_proj,   xproj_h,   XPROJ_N * D_INNER },
    { dt_proj,  dtproj_h,  D_INNER * DT_RANK },
    { out_proj, outproj_h, D_MODEL * D_INNER },
    { ff_w1,    ffw1_h,    4 * D_MODEL * D_MODEL },
    { ff_w2,    ffw2_h,    D_MODEL * 4 * D_MODEL },
  };
  for (int i = 0; i < 6; ++i)
    cvt_f16_kernel<<<(cv[i].n + 255) / 256, blk, 0, stream>>>(cv[i].s, cv[i].d, cv[i].n);

  // LN1: x -> xn_h (f16 only; consumed solely by in_proj GEMM)
  ln_kernel<<<NTOK, blk, 0, stream>>>(x, nullptr, 0, n1w, n1b, nullptr, xn_h);
  // in_proj: xz = xn @ W^T   (N=2048, K=512) -> f32
  gemm_wmma_kernel<ACT_NONE, false, false, true, false>
      <<<dim3(2 * D_INNER / 64, mt), blk, SMEM_B, stream>>>(
      xn_h, D_MODEL, inproj_h, nullptr, nullptr, 0,
      xz, nullptr, 2 * D_INNER, D_MODEL);

  float* outd[2] = { outf, outb };
  for (int dir = 0; dir < 2; ++dir) {
    conv_silu_kernel<<<(NTOK * D_INNER) / 256, blk, 0, stream>>>(
        xz, conv_w, conv_b, xs, xs_h, dir);
    // x_proj: dbl = xs @ W^T (N=64, K=1024) -> f32 (scan) + f16 (dt_proj A)
    gemm_wmma_kernel<ACT_NONE, false, false, true, true>
        <<<dim3(XPROJ_N / 64, mt), blk, SMEM_B, stream>>>(
        xs_h, D_INNER, xproj_h, nullptr, nullptr, 0,
        dbl, dbl_h, XPROJ_N, D_INNER);
    // dt_proj: del = softplus(dbl[:, :32] @ W^T + b) -> f32
    gemm_wmma_kernel<ACT_SOFTPLUS, true, false, true, false>
        <<<dim3(D_INNER / 64, mt), blk, SMEM_B, stream>>>(
        dbl_h, XPROJ_N, dtproj_h, dt_b, nullptr, 0,
        del, nullptr, D_INNER, DT_RANK);
    // selective scan + D-skip + silu(z) gate -> f16
    scan_gate_kernel<<<(BATCH * D_INNER) / 256, blk, 0, stream>>>(
        xs, dbl, del, xz, A_log, Dsk, yg_h, dir);
    // out_proj: outd = yg @ W^T (N=512, K=1024) -> f32
    gemm_wmma_kernel<ACT_NONE, false, false, true, false>
        <<<dim3(D_MODEL / 64, mt), blk, SMEM_B, stream>>>(
        yg_h, D_INNER, outproj_h, nullptr, nullptr, 0,
        outd[dir], nullptr, D_MODEL, D_INNER);
  }

  // LN2 of (out_f + flip(out_b)) -> m_h (f16 only)
  ln_kernel<<<NTOK, blk, 0, stream>>>(outf, outb, 1, n2w, n2b, nullptr, m_h);
  // FFN: ffb = gelu(m @ W1^T + b1)   (N=2048, K=512) -> f16 only
  gemm_wmma_kernel<ACT_GELU, true, false, false, true>
      <<<dim3(4 * D_MODEL / 64, mt), blk, SMEM_B, stream>>>(
      m_h, D_MODEL, ffw1_h, ff_b1, nullptr, 0,
      nullptr, ffb_h, 4 * D_MODEL, D_MODEL);
  // out = ffb @ W2^T + b2 + x   (N=512, K=2048), residual fused -> f32
  gemm_wmma_kernel<ACT_NONE, true, true, true, false>
      <<<dim3(D_MODEL / 64, mt), blk, SMEM_B, stream>>>(
      ffb_h, 4 * D_MODEL, ffw2_h, ff_b2, x, D_MODEL,
      out, nullptr, D_MODEL, 4 * D_MODEL);
}
